// MultiheadSelfAttention_67044439491018
// MI455X (gfx1250) — matrix operand order repaired and checked by reading, not compiled, not run on hardware
//
#include <hip/hip_runtime.h>

typedef __attribute__((ext_vector_type(16))) _Float16 v16h;
typedef __attribute__((ext_vector_type(8)))  _Float16 v8h;
typedef __attribute__((ext_vector_type(4)))  _Float16 v4h;
typedef __attribute__((ext_vector_type(8)))  float    v8f;

#define S_LEN   2048
#define DMODEL  1024
#define HEADS   16
#define DK      64
#define BATCH   2
#define MROWS   (BATCH * S_LEN)   // 4096

static __device__ __forceinline__ v16h join8(v8h a, v8h b) {
  return __builtin_shufflevector(a, b, 0,1,2,3,4,5,6,7,8,9,10,11,12,13,14,15);
}
static __device__ __forceinline__ v8f wmma_f16(v16h a, v16h b, v8f c) {
  // D = A(16x32 f16) * B(32x16 f16) + C(16x16 f32)
  return __builtin_amdgcn_wmma_f32_16x16x32_f16(false, a, false, b, (short)0, c, false, false);
}
static __device__ __forceinline__ v8f zero8() {
  v8f z;
#pragma unroll
  for (int i = 0; i < 8; ++i) z[i] = 0.0f;
  return z;
}

// ---------------------------------------------------------------- converts
__global__ __launch_bounds__(256) void cvt_f32_to_f16(const float* __restrict__ in,
                                                      _Float16* __restrict__ out, int n4) {
  int i = blockIdx.x * blockDim.x + threadIdx.x;
  if (i < n4) {
    float4 v = ((const float4*)in)[i];
    v4h o;
    o[0] = (_Float16)v.x; o[1] = (_Float16)v.y;
    o[2] = (_Float16)v.z; o[3] = (_Float16)v.w;
    ((v4h*)out)[i] = o;
  }
}

// ---------------------------------------------------------------- GEMM: C = A(MxK f16) * W(NxK f16)^T
// Wave tile: 32(M) x 64(N): 2 M-subtiles x 4 N-subtiles, 8 WMMA per K-step,
// 12 b128 loads per K-step (A frags reused 4x, B frags reused 2x).
// Block = 8 waves -> 32(M) x 512(N). Grid = (N/512, M/32).
// mode 0: f16 head-split (b,h,s,dk)   mode 1: f16 transposed head-split (b,h,dk,s)
// mode 2: f32 plain (M x N)
__global__ __launch_bounds__(256) void gemm_xwT(const _Float16* __restrict__ A,
                                                const _Float16* __restrict__ W,
                                                void* __restrict__ dst, int mode) {
  const int lane = threadIdx.x & 31;
  const int wave = threadIdx.x >> 5;
  const int h16  = lane >> 4;          // which lane-half
  const int l    = lane & 15;
  const int m0   = blockIdx.y * 32;
  const int nb   = blockIdx.x * 512 + wave * 64;

  v8f acc[2][4];
#pragma unroll
  for (int r = 0; r < 2; ++r)
#pragma unroll
    for (int t = 0; t < 4; ++t) acc[r][t] = zero8();

  const _Float16* arow0 = A + (size_t)(m0 + l) * DMODEL;
  const _Float16* arow1 = A + (size_t)(m0 + 16 + l) * DMODEL;

  for (int k0 = 0; k0 < DMODEL; k0 += 32) {
    // A fragments: lane<16 -> K {k0..k0+7, k0+16..k0+23}; lane>=16 -> +8
    v16h af[2];
    {
      v8h a0 = *(const v8h*)(arow0 + k0 + 8 * h16);
      v8h a1 = *(const v8h*)(arow0 + k0 + 16 + 8 * h16);
      af[0] = join8(a0, a1);
      v8h b0 = *(const v8h*)(arow1 + k0 + 8 * h16);
      v8h b1 = *(const v8h*)(arow1 + k0 + 16 + 8 * h16);
      af[1] = join8(b0, b1);
    }
#pragma unroll
    for (int t = 0; t < 4; ++t) {
      // B fragment: lane n holds column (nb+16t+n), K {k0..k0+7, k0+16..k0+23} (+8 for lane>=16)
      const _Float16* wrow = W + (size_t)(nb + 16 * t + l) * DMODEL + k0 + 8 * h16;
      v16h bf = join8(*(const v8h*)wrow, *(const v8h*)(wrow + 16));
      acc[0][t] = wmma_f16(af[0], bf, acc[0][t]);
      acc[1][t] = wmma_f16(af[1], bf, acc[1][t]);
    }
  }

  if (mode == 2) {
    float* o = (float*)dst;
#pragma unroll
    for (int r = 0; r < 2; ++r)
#pragma unroll
      for (int t = 0; t < 4; ++t)
#pragma unroll
        for (int j = 0; j < 8; ++j) {
          int m = m0 + 16 * r + j + 8 * h16;
          int n = nb + 16 * t + l;
          o[(size_t)m * DMODEL + n] = acc[r][t][j];
        }
  } else {
    _Float16* o = (_Float16*)dst;
#pragma unroll
    for (int r = 0; r < 2; ++r)
#pragma unroll
      for (int t = 0; t < 4; ++t)
#pragma unroll
        for (int j = 0; j < 8; ++j) {
          int m = m0 + 16 * r + j + 8 * h16;
          int n = nb + 16 * t + l;
          int b = m >> 11, s = m & (S_LEN - 1);
          int h = n >> 6,  d = n & (DK - 1);
          size_t idx = (mode == 0)
                     ? (((size_t)(b * HEADS + h) * S_LEN + s) * DK + d)    // (b,h,s,dk)
                     : (((size_t)(b * HEADS + h) * DK + d) * S_LEN + s);   // (b,h,dk,s)
          o[idx] = (_Float16)acc[r][t][j];
        }
  }
}

// ---------------------------------------------------------------- flash attention (causal)
// One wave per 32 queries (two 16-query column tiles). Scores computed
// transposed (S^T = K * Q^T) so softmax is lane-local (+ one shfl_xor(16))
// and the O^T rescale is a per-lane scalar. Key blocks of 32 keys.
// Every K fragment and V fragment is reused by both query tiles:
// 16 b128 loads per 16 WMMAs per key block.
__global__ __launch_bounds__(256) void attn_kernel(const _Float16* __restrict__ Qh,
                                                   const _Float16* __restrict__ Kh,
                                                   const _Float16* __restrict__ VT,
                                                   _Float16* __restrict__ AO) {
  const int lane = threadIdx.x & 31;
  const int wave = threadIdx.x >> 5;
  const int h16  = lane >> 4;
  const int l    = lane & 15;
  const int wid  = blockIdx.x * 8 + wave;
  const int qt   = wid & (S_LEN / 32 - 1);   // 0..63
  const int bh   = wid >> 6;                 // 0..31
  const int q0   = qt * 32;

  const _Float16* Qb = Qh + (size_t)bh * S_LEN * DK;
  const _Float16* Kb = Kh + (size_t)bh * S_LEN * DK;
  const _Float16* Vb = VT + (size_t)bh * DK * S_LEN;

  // Q^T as B fragments (rows = d, cols = query); lane l holds query q0+16c+l
  v16h qb[2][2];
#pragma unroll
  for (int c = 0; c < 2; ++c)
#pragma unroll
    for (int ks = 0; ks < 2; ++ks)
      qb[c][ks] = join8(*(const v8h*)(Qb + (size_t)(q0 + 16 * c + l) * DK + 32 * ks + 8 * h16), *(const v8h*)(Qb + (size_t)(q0 + 16 * c + l) * DK + 32 * ks + 16 + 8 * h16));

  v8f oT[2][4];
#pragma unroll
  for (int c = 0; c < 2; ++c)
#pragma unroll
    for (int t = 0; t < 4; ++t) oT[c][t] = zero8();
  float m_run[2] = { -1e30f, -1e30f };
  float l_run[2] = { 0.0f, 0.0f };
  const int qIdx0 = q0 + l;          // query of tile c, = qIdx0 + 16c
  const int nkb   = (q0 + 31) / 32 + 1;

  for (int kb32 = 0; kb32 < nkb; ++kb32) {
    const int kb = kb32 * 32;
    v8f st[2][2];
#pragma unroll
    for (int c = 0; c < 2; ++c) { st[c][0] = zero8(); st[c][1] = zero8(); }

    // S^T = K * Q^T ; A = K rows (M = key), d(=WMMA K) split into 2 steps
#pragma unroll
    for (int f = 0; f < 2; ++f) {
      const _Float16* krow = Kb + (size_t)(kb + 16 * f + l) * DK;
#pragma unroll
      for (int ks = 0; ks < 2; ++ks) {
        v8h c0 = *(const v8h*)(krow + 32 * ks + 8 * h16);
        v8h c1 = *(const v8h*)(krow + 32 * ks + 16 + 8 * h16);
        v16h kf = join8(c0, c1);
        st[0][f] = wmma_f16(kf, qb[0][ks], st[0][f]);
        st[1][f] = wmma_f16(kf, qb[1][ks], st[1][f]);
      }
    }

    v16h pb[2];
#pragma unroll
    for (int c = 0; c < 2; ++c) {
      const int qIdx = qIdx0 + 16 * c;
      // scale + causal mask (st[c][f][j]: key = kb+16f+j+8*h16)
#pragma unroll
      for (int f = 0; f < 2; ++f)
#pragma unroll
        for (int j = 0; j < 8; ++j) {
          int key = kb + 16 * f + j + 8 * h16;
          float v = st[c][f][j] * 0.125f;   // 1/sqrt(64)
          st[c][f][j] = (key <= qIdx) ? v : -1e30f;
        }
      // online softmax (per-lane = per-query; merge the two lane-halves)
      float bm = -1e30f;
#pragma unroll
      for (int f = 0; f < 2; ++f)
#pragma unroll
        for (int j = 0; j < 8; ++j) bm = fmaxf(bm, st[c][f][j]);
      bm = fmaxf(bm, __shfl_xor(bm, 16, 32));
      float m_new = fmaxf(m_run[c], bm);
      float corr  = __expf(m_run[c] - m_new);
      float lsum  = 0.0f;
#pragma unroll
      for (int f = 0; f < 2; ++f)
#pragma unroll
        for (int j = 0; j < 8; ++j) {
          float p = __expf(st[c][f][j] - m_new);
          st[c][f][j] = p;
          lsum += p;
        }
      lsum += __shfl_xor(lsum, 16, 32);
      l_run[c] = l_run[c] * corr + lsum;
      m_run[c] = m_new;
#pragma unroll
      for (int t = 0; t < 4; ++t)
#pragma unroll
        for (int i = 0; i < 8; ++i) oT[c][t][i] *= corr;

      // Rearrange P^T (C-layout) into B-fragment layout
      // (lane n: keys 0..15; lane n+16: keys 16..31)
#pragma unroll
      for (int v = 0; v < 4; ++v) {
        float x0 = st[c][0][2 * v],     x1 = st[c][0][2 * v + 1];
        float y0 = __shfl_xor(st[c][1][2 * v], 16, 32);
        float y1 = __shfl_xor(st[c][1][2 * v + 1], 16, 32);
        pb[c][2 * v]     = (_Float16)(h16 ? y0 : x0);
        pb[c][2 * v + 1] = (_Float16)(h16 ? y1 : x1);
      }
#pragma unroll
      for (int v = 4; v < 8; ++v) {
        int j = 2 * v - 8;
        float x0 = __shfl_xor(st[c][0][j], 16, 32);
        float x1 = __shfl_xor(st[c][0][j + 1], 16, 32);
        float y0 = st[c][1][j], y1 = st[c][1][j + 1];
        pb[c][2 * v]     = (_Float16)(h16 ? y0 : x0);
        pb[c][2 * v + 1] = (_Float16)(h16 ? y1 : x1);
      }
    }

    // O^T += V^T * P^T   (A = V^T rows = d, contiguous; reused by both tiles)
#pragma unroll
    for (int mt = 0; mt < 4; ++mt) {
      const _Float16* vrow = Vb + (size_t)(16 * mt + l) * S_LEN + kb;
      v16h vf = *(const v16h*)(vrow + 16 * h16);
      oT[0][mt] = wmma_f16(vf, pb[0], oT[0][mt]);
      oT[1][mt] = wmma_f16(vf, pb[1], oT[1][mt]);
    }
  }

  const int b = bh >> 4, h = bh & (HEADS - 1);
#pragma unroll
  for (int c = 0; c < 2; ++c) {
    const float inv_l = 1.0f / l_run[c];
    const int s = q0 + 16 * c + l;
#pragma unroll
    for (int mt = 0; mt < 4; ++mt)
#pragma unroll
      for (int j = 0; j < 8; ++j) {
        int d = 16 * mt + j + 8 * h16;
        AO[((size_t)b * S_LEN + s) * DMODEL + h * DK + d] = (_Float16)(oT[c][mt][j] * inv_l);
      }
  }
}

// ---------------------------------------------------------------- launcher
extern "C" void kernel_launch(void* const* d_in, const int* in_sizes, int n_in,
                              void* d_out, int out_size, void* d_ws, size_t ws_size,
                              hipStream_t stream) {
  const float* x  = (const float*)d_in[0];
  const float* wq = (const float*)d_in[1];
  const float* wk = (const float*)d_in[2];
  const float* wv = (const float*)d_in[3];
  const float* wo = (const float*)d_in[4];
  float* out = (float*)d_out;

  _Float16* ws = (_Float16*)d_ws;
  size_t off = 0;
  _Float16* xh  = ws + off; off += (size_t)MROWS * DMODEL;            // 4M
  _Float16* wqh = ws + off; off += (size_t)DMODEL * DMODEL;           // 1M
  _Float16* wkh = ws + off; off += (size_t)DMODEL * DMODEL;
  _Float16* wvh = ws + off; off += (size_t)DMODEL * DMODEL;
  _Float16* woh = ws + off; off += (size_t)DMODEL * DMODEL;
  _Float16* Qh  = ws + off; off += (size_t)BATCH * HEADS * S_LEN * DK; // 4M
  _Float16* Kh  = ws + off; off += (size_t)BATCH * HEADS * S_LEN * DK;
  _Float16* VTh = ws + off; off += (size_t)BATCH * HEADS * DK * S_LEN;
  _Float16* AOh = ws + off; off += (size_t)MROWS * DMODEL;
  // total: 24M halves = 48 MB

  // 1) fp32 -> fp16 conversions
  cvt_f32_to_f16<<<(MROWS * DMODEL / 4 + 255) / 256, 256, 0, stream>>>(x, xh, MROWS * DMODEL / 4);
  cvt_f32_to_f16<<<(DMODEL * DMODEL / 4 + 255) / 256, 256, 0, stream>>>(wq, wqh, DMODEL * DMODEL / 4);
  cvt_f32_to_f16<<<(DMODEL * DMODEL / 4 + 255) / 256, 256, 0, stream>>>(wk, wkh, DMODEL * DMODEL / 4);
  cvt_f32_to_f16<<<(DMODEL * DMODEL / 4 + 255) / 256, 256, 0, stream>>>(wv, wvh, DMODEL * DMODEL / 4);
  cvt_f32_to_f16<<<(DMODEL * DMODEL / 4 + 255) / 256, 256, 0, stream>>>(wo, woh, DMODEL * DMODEL / 4);

  // 2) projections: Q,K head-split; V transposed per head
  dim3 g(DMODEL / 512, MROWS / 32);
  gemm_xwT<<<g, 256, 0, stream>>>(xh, wqh, (void*)Qh, 0);
  gemm_xwT<<<g, 256, 0, stream>>>(xh, wkh, (void*)Kh, 0);
  gemm_xwT<<<g, 256, 0, stream>>>(xh, wvh, (void*)VTh, 1);

  // 3) causal flash attention: 2048 query tiles of 32, 8 waves per block
  attn_kernel<<<(BATCH * HEADS * (S_LEN / 32)) / 8, 256, 0, stream>>>(Qh, Kh, VTh, AOh);

  // 4) output projection -> fp32
  gemm_xwT<<<g, 256, 0, stream>>>(AOh, woh, (void*)out, 2);
}
